// BiAttention_26766236189054
// MI455X (gfx1250) — matrix operand order repaired and checked by reading, compiled, not run on hardware
//
#include <hip/hip_runtime.h>

// ---------------- types ----------------
typedef __bf16 bf16_t;
typedef __attribute__((ext_vector_type(8)))  __bf16 v8bf;
typedef __attribute__((ext_vector_type(16))) __bf16 v16bf;
typedef __attribute__((ext_vector_type(8)))  float  v8f;

#define BB 32
#define CC 256
#define QQ 128
#define DD 1024
#define NEG_INF (-3.402823466e38f)

// d_output is (B, 4C=1024, D); q_output is (B, 4C=1024, Q), concatenated flat.
#define QOUT_OFF ((size_t)BB * 1024 * DD)

// round-to-nearest-even f32 -> bf16
__device__ inline bf16_t f2bf(float x) {
    unsigned u = __float_as_uint(x);
    unsigned short s = (unsigned short)((u + 0x7FFFu + ((u >> 16) & 1u)) >> 16);
    return __builtin_bit_cast(__bf16, s);
}

__device__ inline float wave_max(float v) {
#pragma unroll
    for (int o = 16; o; o >>= 1) v = fmaxf(v, __shfl_xor(v, o, 32));
    return v;
}
__device__ inline float wave_sum(float v) {
#pragma unroll
    for (int o = 16; o; o >>= 1) v += __shfl_xor(v, o, 32);
    return v;
}

// ---------------- WMMA fragment loads (CDNA5 16x16x32 bf16 layouts) ----------------
// A: 16(M) x 32(K), row-major [m][k]. lane<16: M=lane, K 0-7 (v0-3), 16-23 (v4-7);
// lane>=16: K 8-15 / 24-31.
__device__ inline v16bf load_a_frag(const bf16_t* A, int lda, int m0, int k0, int lane) {
    int m  = m0 + (lane & 15);
    int ks = (lane >> 4) * 8;
    const bf16_t* p = A + (size_t)m * lda + k0 + ks;
    v8bf lo = *(const v8bf*)p;
    v8bf hi = *(const v8bf*)(p + 16);
    return __builtin_shufflevector(lo, hi, 0,1,2,3,4,5,6,7,8,9,10,11,12,13,14,15);
}
// B: 32(K) x 16(N), stored column-major Bc[n][k] (contiguous in k).
// lane<16: N=lane, K 0-7 / 16-23; lane>=16: N=lane-16, K 8-15 / 24-31.
__device__ inline v16bf load_b_frag(const bf16_t* Bc, int ldb, int n0, int k0, int lane) {
    int n  = n0 + (lane & 15);
    int kb = (lane >> 4) * 8;
    const bf16_t* p = Bc + (size_t)n * ldb + k0 + kb;
    v8bf lo = *(const v8bf*)p;
    v8bf hi = *(const v8bf*)(p + 16);
    return __builtin_shufflevector(lo, hi, 0,1,2,3,4,5,6,7,8,9,10,11,12,13,14,15);
}

__device__ inline v8f wmma_bf16(v16bf a, v16bf b, v8f c) {
    return __builtin_amdgcn_wmma_f32_16x16x32_bf16(false, a, false, b, (short)0, c, false, false);
}

// 32x32 output per wave: 2x2 WMMA tiles, 2 loads per WMMA instead of 4.
struct Acc4 { v8f a00, a01, a10, a11; };
__device__ inline Acc4 gemm_tile_2x2(const bf16_t* A, int lda, const bf16_t* Bc, int ldb,
                                     int m0, int n0, int K, int lane) {
    Acc4 acc;
    acc.a00 = (v8f){}; acc.a01 = (v8f){}; acc.a10 = (v8f){}; acc.a11 = (v8f){};
    for (int k0 = 0; k0 < K; k0 += 32) {
        v16bf a0 = load_a_frag(A, lda, m0,      k0, lane);
        v16bf a1 = load_a_frag(A, lda, m0 + 16, k0, lane);
        v16bf b0 = load_b_frag(Bc, ldb, n0,      k0, lane);
        v16bf b1 = load_b_frag(Bc, ldb, n0 + 16, k0, lane);
        acc.a00 = wmma_bf16(a0, b0, acc.a00);
        acc.a01 = wmma_bf16(a0, b1, acc.a01);
        acc.a10 = wmma_bf16(a1, b0, acc.a10);
        acc.a11 = wmma_bf16(a1, b1, acc.a11);
    }
    return acc;
}
// C/D layout: lane holds column N=lane&15; rows = tile_m0 + 8*(lane>>4) + r, r=0..7.

// ---------------- prep ----------------
__global__ void prep_doc_kernel(const float* __restrict__ doc, bf16_t* doc_bf,
                                bf16_t* docT, float* d_out) {
    size_t i = (size_t)blockIdx.x * 256 + threadIdx.x;   // B*C*D threads
    int d = (int)(i % DD);
    int c = (int)((i / DD) % CC);
    int b = (int)(i / ((size_t)DD * CC));
    float v = doc[i];
    doc_bf[i] = f2bf(v);
    docT[((size_t)b * DD + d) * CC + c] = f2bf(v);
    d_out[(size_t)b * 1024 * DD + (size_t)c * DD + d] = v;   // rows 0..255 = doc
}

__global__ void prep_query_kernel(const float* __restrict__ query, const float* __restrict__ W,
                                  bf16_t* query_bf, bf16_t* Aq, float* a_q, float* q_out) {
    int b = blockIdx.x;
    int q = threadIdx.x;                                  // 128 threads
    const float* Qb = query + (size_t)b * CC * QQ;
    float aq = 0.f;
    for (int c = 0; c < CC; ++c) {
        float v = Qb[(size_t)c * QQ + q];
        aq += W[CC + c] * v;
        query_bf[(size_t)b * CC * QQ + (size_t)c * QQ + q] = f2bf(v);
        Aq[((size_t)b * QQ + q) * CC + c] = f2bf(W[2 * CC + c] * v + W[c]); // folds a_d
        q_out[(size_t)b * 1024 * QQ + (size_t)c * QQ + q] = v;  // rows 0..255 = query
    }
    a_q[b * QQ + q] = aq;
}

// ---------------- G1: sim = Aq(QxC) @ doc(CxD) + a_q ----------------
__global__ void g1_sim_kernel(const bf16_t* __restrict__ Aq, const bf16_t* __restrict__ docT,
                              const float* __restrict__ a_q, float* sim) {
    int b = blockIdx.y;
    int wave = threadIdx.x >> 5, lane = threadIdx.x & 31;
    int tile = blockIdx.x * 8 + wave;          // 128 units: Mt2=4, Nt2=32
    int m0 = (tile & 3) * 32;
    int n0 = (tile >> 2) * 32;
    Acc4 acc = gemm_tile_2x2(Aq + (size_t)b * QQ * CC, CC,
                             docT + (size_t)b * DD * CC, CC, m0, n0, CC, lane);
    float* S = sim + (size_t)b * QQ * DD;
    const float* aq = a_q + b * QQ;
    v8f av[2][2] = {{acc.a00, acc.a01}, {acc.a10, acc.a11}};
#pragma unroll
    for (int i = 0; i < 2; ++i) {
        int mb = m0 + 16 * i + 8 * (lane >> 4);
#pragma unroll
        for (int j = 0; j < 2; ++j) {
            int n = n0 + 16 * j + (lane & 15);
#pragma unroll
            for (int r = 0; r < 8; ++r)
                S[(size_t)(mb + r) * DD + n] = av[i][j][r] + aq[mb + r];
        }
    }
}

// ---------------- softmax over d (rows): s2[q][d], bf16 ----------------
__global__ void softmax_d_kernel(const float* __restrict__ sim, bf16_t* s2) {
    int b = blockIdx.y, q = blockIdx.x;
    const float* row = sim + ((size_t)b * QQ + q) * DD;
    __shared__ float sm[8];
    int t = threadIdx.x, w = t >> 5, lane = t & 31;
    float x[4], m = NEG_INF;
#pragma unroll
    for (int i = 0; i < 4; ++i) { x[i] = row[t + i * 256]; m = fmaxf(m, x[i]); }
    m = wave_max(m);
    if (lane == 0) sm[w] = m;
    __syncthreads();
    float bm = sm[0];
#pragma unroll
    for (int i = 1; i < 8; ++i) bm = fmaxf(bm, sm[i]);
    __syncthreads();
    float e[4], s = 0.f;
#pragma unroll
    for (int i = 0; i < 4; ++i) { e[i] = __expf(x[i] - bm); s += e[i]; }
    s = wave_sum(s);
    if (lane == 0) sm[w] = s;
    __syncthreads();
    float bs = 0.f;
#pragma unroll
    for (int i = 0; i < 8; ++i) bs += sm[i];
    float inv = 1.0f / bs;
    bf16_t* o = s2 + ((size_t)b * QQ + q) * DD;
#pragma unroll
    for (int i = 0; i < 4; ++i) o[t + i * 256] = f2bf(e[i] * inv);
}

// ---------------- softmax over q (cols): s1T[d][q], bf16 ----------------
__global__ void softmax_q_kernel(const float* __restrict__ sim, bf16_t* s1T) {
    int b = blockIdx.y;
    int d = blockIdx.x * 256 + threadIdx.x;
    const float* S = sim + (size_t)b * QQ * DD;
    float m = NEG_INF;
    for (int q = 0; q < QQ; ++q) m = fmaxf(m, S[(size_t)q * DD + d]);
    float s = 0.f;
    for (int q = 0; q < QQ; ++q) s += __expf(S[(size_t)q * DD + d] - m);
    float inv = 1.0f / s;
    bf16_t* o = s1T + ((size_t)b * DD + d) * QQ;
    for (int q = 0; q < QQ; ++q) o[q] = f2bf(__expf(S[(size_t)q * DD + d] - m) * inv);
}

// ---------------- G2: d_d2q = query(CxQ) @ s1(QxD); fused epilogue ----------------
__global__ void g2_dd2q_kernel(const bf16_t* __restrict__ query_bf, const bf16_t* __restrict__ s1T,
                               const float* __restrict__ doc, float* d_out, bf16_t* dd2q_bf) {
    int b = blockIdx.y;
    int wave = threadIdx.x >> 5, lane = threadIdx.x & 31;
    int tile = blockIdx.x * 8 + wave;          // 256 units: Mt2=8, Nt2=32
    int m0 = (tile & 7) * 32;
    int n0 = (tile >> 3) * 32;
    Acc4 acc = gemm_tile_2x2(query_bf + (size_t)b * CC * QQ, QQ,
                             s1T + (size_t)b * DD * QQ, QQ, m0, n0, QQ, lane);
    float* Do = d_out + (size_t)b * 1024 * DD;
    const float* dc = doc + (size_t)b * CC * DD;
    bf16_t* DB = dd2q_bf + (size_t)b * CC * DD;
    v8f av[2][2] = {{acc.a00, acc.a01}, {acc.a10, acc.a11}};
#pragma unroll
    for (int i = 0; i < 2; ++i) {
        int mb = m0 + 16 * i + 8 * (lane >> 4);
#pragma unroll
        for (int j = 0; j < 2; ++j) {
            int n = n0 + 16 * j + (lane & 15);
#pragma unroll
            for (int r = 0; r < 8; ++r) {
                int m = mb + r;
                float v  = av[i][j][r];
                float dv = dc[(size_t)m * DD + n];
                Do[(size_t)(256 + m) * DD + n] = v;        // d_d2q
                Do[(size_t)(512 + m) * DD + n] = dv * v;   // doc * d_d2q
                DB[(size_t)m * DD + n] = f2bf(v);
            }
        }
    }
}

// ---------------- G3: q_q2d = doc(CxD) @ s2^T(DxQ); fused epilogue ----------------
__global__ void g3_qq2d_kernel(const bf16_t* __restrict__ doc_bf, const bf16_t* __restrict__ s2,
                               const float* __restrict__ query, float* q_out, bf16_t* qq2d_bf) {
    int b = blockIdx.y;
    int wave = threadIdx.x >> 5, lane = threadIdx.x & 31;
    int tile = blockIdx.x * 8 + wave;          // 32 units: Mt2=8, Nt2=4
    int m0 = (tile & 7) * 32;
    int n0 = (tile >> 3) * 32;
    Acc4 acc = gemm_tile_2x2(doc_bf + (size_t)b * CC * DD, DD,
                             s2 + (size_t)b * QQ * DD, DD, m0, n0, DD, lane);
    float* Qo = q_out + (size_t)b * 1024 * QQ;
    const float* qc = query + (size_t)b * CC * QQ;
    bf16_t* QB = qq2d_bf + (size_t)b * CC * QQ;
    v8f av[2][2] = {{acc.a00, acc.a01}, {acc.a10, acc.a11}};
#pragma unroll
    for (int i = 0; i < 2; ++i) {
        int mb = m0 + 16 * i + 8 * (lane >> 4);
#pragma unroll
        for (int j = 0; j < 2; ++j) {
            int n = n0 + 16 * j + (lane & 15);
#pragma unroll
            for (int r = 0; r < 8; ++r) {
                int m = mb + r;
                float v  = av[i][j][r];
                float qv = qc[(size_t)m * QQ + n];
                Qo[(size_t)(256 + m) * QQ + n] = v;        // q_q2d
                Qo[(size_t)(512 + m) * QQ + n] = qv * v;   // query * q_q2d
                QB[(size_t)m * QQ + n] = f2bf(v);
            }
        }
    }
}

// ---------------- G4: d_q2d = q_q2d(CxQ) @ s1(QxD); writes doc*d_q2d ----------------
__global__ void g4_dq2d_kernel(const bf16_t* __restrict__ qq2d_bf, const bf16_t* __restrict__ s1T,
                               const float* __restrict__ doc, float* d_out) {
    int b = blockIdx.y;
    int wave = threadIdx.x >> 5, lane = threadIdx.x & 31;
    int tile = blockIdx.x * 8 + wave;          // 256 units
    int m0 = (tile & 7) * 32;
    int n0 = (tile >> 3) * 32;
    Acc4 acc = gemm_tile_2x2(qq2d_bf + (size_t)b * CC * QQ, QQ,
                             s1T + (size_t)b * DD * QQ, QQ, m0, n0, QQ, lane);
    float* Do = d_out + (size_t)b * 1024 * DD;
    const float* dc = doc + (size_t)b * CC * DD;
    v8f av[2][2] = {{acc.a00, acc.a01}, {acc.a10, acc.a11}};
#pragma unroll
    for (int i = 0; i < 2; ++i) {
        int mb = m0 + 16 * i + 8 * (lane >> 4);
#pragma unroll
        for (int j = 0; j < 2; ++j) {
            int n = n0 + 16 * j + (lane & 15);
#pragma unroll
            for (int r = 0; r < 8; ++r) {
                int m = mb + r;
                Do[(size_t)(768 + m) * DD + n] = dc[(size_t)m * DD + n] * av[i][j][r];
            }
        }
    }
}

// ---------------- G5: q_d2q = d_d2q(CxD) @ s2^T(DxQ); writes query*q_d2q ----------------
__global__ void g5_qd2q_kernel(const bf16_t* __restrict__ dd2q_bf, const bf16_t* __restrict__ s2,
                               const float* __restrict__ query, float* q_out) {
    int b = blockIdx.y;
    int wave = threadIdx.x >> 5, lane = threadIdx.x & 31;
    int tile = blockIdx.x * 8 + wave;          // 32 units
    int m0 = (tile & 7) * 32;
    int n0 = (tile >> 3) * 32;
    Acc4 acc = gemm_tile_2x2(dd2q_bf + (size_t)b * CC * DD, DD,
                             s2 + (size_t)b * QQ * DD, DD, m0, n0, DD, lane);
    float* Qo = q_out + (size_t)b * 1024 * QQ;
    const float* qc = query + (size_t)b * CC * QQ;
    v8f av[2][2] = {{acc.a00, acc.a01}, {acc.a10, acc.a11}};
#pragma unroll
    for (int i = 0; i < 2; ++i) {
        int mb = m0 + 16 * i + 8 * (lane >> 4);
#pragma unroll
        for (int j = 0; j < 2; ++j) {
            int n = n0 + 16 * j + (lane & 15);
#pragma unroll
            for (int r = 0; r < 8; ++r) {
                int m = mb + r;
                Qo[(size_t)(768 + m) * QQ + n] = qc[(size_t)m * QQ + n] * av[i][j][r];
            }
        }
    }
}

// ---------------- host launch ----------------
extern "C" void kernel_launch(void* const* d_in, const int* in_sizes, int n_in,
                              void* d_out, int out_size, void* d_ws, size_t ws_size,
                              hipStream_t stream) {
    const float* doc   = (const float*)d_in[0];
    const float* query = (const float*)d_in[1];
    const float* W     = (const float*)d_in[2];
    float* out = (float*)d_out;
    float* q_out = out + QOUT_OFF;

    char* ws = (char*)d_ws;
    size_t off = 0;
    float* sim = (float*)(ws + off);        off += (size_t)BB * QQ * DD * 4;   // 16 MB
    float* a_q = (float*)(ws + off);        off += (size_t)BB * QQ * 4;
    bf16_t* Aq       = (bf16_t*)(ws + off); off += (size_t)BB * QQ * CC * 2;
    bf16_t* query_bf = (bf16_t*)(ws + off); off += (size_t)BB * CC * QQ * 2;
    bf16_t* doc_bf   = (bf16_t*)(ws + off); off += (size_t)BB * CC * DD * 2;   // 16 MB
    bf16_t* docT     = (bf16_t*)(ws + off); off += (size_t)BB * DD * CC * 2;   // 16 MB
    bf16_t* s1T      = (bf16_t*)(ws + off); off += (size_t)BB * DD * QQ * 2;   // 8 MB
    bf16_t* s2       = (bf16_t*)(ws + off); off += (size_t)BB * QQ * DD * 2;   // 8 MB
    bf16_t* dd2q_bf  = (bf16_t*)(ws + off); off += (size_t)BB * CC * DD * 2;   // 16 MB
    bf16_t* qq2d_bf  = (bf16_t*)(ws + off); off += (size_t)BB * CC * QQ * 2;
    (void)ws_size; (void)in_sizes; (void)n_in; (void)out_size;

    // prep
    prep_doc_kernel<<<(BB * CC * DD) / 256, 256, 0, stream>>>(doc, doc_bf, docT, out);
    prep_query_kernel<<<BB, QQ, 0, stream>>>(query, W, query_bf, Aq, a_q, q_out);
    // G1: sim (M=128, N=1024, K=256) -> 128 wave-units of 32x32, 8 waves/block
    g1_sim_kernel<<<dim3(16, BB), 256, 0, stream>>>(Aq, docT, a_q, sim);
    // softmaxes
    softmax_d_kernel<<<dim3(QQ, BB), 256, 0, stream>>>(sim, s2);
    softmax_q_kernel<<<dim3(DD / 256, BB), 256, 0, stream>>>(sim, s1T);
    // G2: (M=256, N=1024, K=128) -> 256 wave-units
    g2_dd2q_kernel<<<dim3(32, BB), 256, 0, stream>>>(query_bf, s1T, doc, out, dd2q_bf);
    // G3: (M=256, N=128, K=1024) -> 32 wave-units
    g3_qq2d_kernel<<<dim3(4, BB), 256, 0, stream>>>(doc_bf, s2, query, q_out, qq2d_bf);
    // G4: (M=256, N=1024, K=128)
    g4_dq2d_kernel<<<dim3(32, BB), 256, 0, stream>>>(qq2d_bf, s1T, doc, out);
    // G5: (M=256, N=128, K=1024)
    g5_qd2q_kernel<<<dim3(4, BB), 256, 0, stream>>>(dd2q_bf, s2, query, q_out);
}
